// QuantumKernelAttention_54417235640463
// MI455X (gfx1250) — matrix operand order repaired and checked by reading, not compiled, not run on hardware
//
#include <hip/hip_runtime.h>

// ---------------------------------------------------------------------------
// QuantumKernelAttention for MI455X (gfx1250, wave32, WMMA, async->LDS).
//   K1 encode    : angles + 6-qubit circuit (2 VGPRs x 32 lanes, shfl_xor);
//                  writes f32 q/k states + bf16 copy of x.
//   K2 convert   : Wv, Wo -> bf16 (once).
//   K3 gemm      : vT_bf[b][n][s] = bf16(x@Wv.T+bv)            (bf16 WMMA)
//   K4 scores    : overlap^2 -> centered -> *exp(ls) -> mask   (f32 WMMA 16x16x4)
//   K5 softmax   : row softmax -> f32 attention output + bf16 copy
//   K6 gemm      : attnV_bf = attention @ v  (Bt = vT_bf/batch)(bf16 WMMA)
//   K7 gemm      : out = attnV @ Wo.T + bo  (f32 out)          (bf16 WMMA)
// All GEMM LDS staging is a pure byte copy -> GLOBAL_LOAD_ASYNC_TO_LDS_B128
// (ASYNCcnt) with double buffering in the K-loop.
// ---------------------------------------------------------------------------

typedef __attribute__((ext_vector_type(2)))  float  v2f;
typedef __attribute__((ext_vector_type(8)))  float  v8f;
typedef __attribute__((ext_vector_type(16))) __bf16 v16bf;
typedef __attribute__((ext_vector_type(8)))  __bf16 v8bf;
typedef int v4i_g __attribute__((vector_size(16)));   // matches async builtin pointee

union FragBF { v16bf v; v8bf h[2]; };

constexpr int BATCH = 8;
constexpr int SEQ   = 2048;
constexpr int DM    = 1024;
constexpr int DST   = 64;

#if defined(__HIP_DEVICE_COMPILE__) && defined(__has_builtin)
#  if __has_builtin(__builtin_amdgcn_global_load_async_to_lds_b128)
#    define HAVE_ASYNC_B128 1
#  endif
#endif

__device__ __forceinline__ unsigned short f2bf(float f) {
  unsigned u = __float_as_uint(f);
  u += 0x7FFFu + ((u >> 16) & 1u);          // round-to-nearest-even
  return (unsigned short)(u >> 16);
}

// 16-byte global -> LDS copy (async when available, ASYNCcnt-tracked).
__device__ __forceinline__ void g2l_16B(void* lds, const void* g) {
#if defined(HAVE_ASYNC_B128)
  __builtin_amdgcn_global_load_async_to_lds_b128(
      (__attribute__((address_space(1))) v4i_g*)g,
      (__attribute__((address_space(3))) v4i_g*)lds, 0, 0);
#else
  *(uint4*)lds = *(const uint4*)g;
#endif
}

__device__ __forceinline__ void async_fence() {
#if defined(HAVE_ASYNC_B128)
#  if __has_builtin(__builtin_amdgcn_s_wait_asynccnt)
  __builtin_amdgcn_s_wait_asynccnt(0);
#  else
  asm volatile("s_wait_asynccnt 0x0" ::: "memory");
#  endif
#endif
}

// ---------------------------------------------------------------------------
// K1: angle projection + quantum encode, one wave32 per token.
// ---------------------------------------------------------------------------
__device__ __forceinline__ void run_circuit(const float* th, float* outp, int lane) {
  float c[6], s[6];
#pragma unroll
  for (int q = 0; q < 6; ++q) { float h = th[q] * 0.5f; c[q] = __cosf(h); s[q] = __sinf(h); }
  float e0 = (lane == 0) ? 1.0f : 0.0f;
  float e1 = 0.0f;
#pragma unroll
  for (int d = 0; d < 3; ++d) {
    { float n0 = c[0]*e0 - s[0]*e1; float n1 = s[0]*e0 + c[0]*e1; e0 = n0; e1 = n1; }
#pragma unroll
    for (int q = 1; q < 6; ++q) {           // RY: qubit bit (5-q) = lane bit
      int str = 32 >> q;
      int bit = lane & str;
      float p0 = __shfl_xor(e0, str, 32);
      float p1 = __shfl_xor(e1, str, 32);
      e0 = bit ? (s[q]*p0 + c[q]*e0) : (c[q]*e0 - s[q]*p0);
      e1 = bit ? (s[q]*p1 + c[q]*e1) : (c[q]*e1 - s[q]*p1);
    }
    e1 = __shfl_xor(e1, 16, 32);            // CNOT q=0 (ctrl bit5 = reg)
#pragma unroll
    for (int q = 1; q < 5; ++q) {           // CNOT: ctrl bit (5-q), tgt (4-q)
      int cm = 32 >> q, tm = 16 >> q;
      float p0 = __shfl_xor(e0, tm, 32);
      float p1 = __shfl_xor(e1, tm, 32);
      if (lane & cm) { e0 = p0; e1 = p1; }
    }
  }
  outp[lane]      = e0;
  outp[32 + lane] = e1;
}

__global__ __launch_bounds__(256) void encode_kernel(
    const float* __restrict__ x,
    const float* __restrict__ Wq, const float* __restrict__ bq,
    const float* __restrict__ Wk, const float* __restrict__ bk,
    float* __restrict__ qs, float* __restrict__ ks,
    unsigned short* __restrict__ xbf) {
  __shared__ __align__(16) float Wsh[12 * DM];
  const int tid = threadIdx.x, lane = tid & 31, w = tid >> 5;

  float4* Wsh4 = (float4*)Wsh;
  const float4* Wq4 = (const float4*)Wq;
  const float4* Wk4 = (const float4*)Wk;
#pragma unroll
  for (int j = 0; j < 12; ++j) {
    int g = tid + 256 * j;
    Wsh4[g] = (g < 1536) ? Wq4[g] : Wk4[g - 1536];
  }
  __syncthreads();

  const size_t t = (size_t)blockIdx.x * 8 + w;
  const float4* xr = (const float4*)(x + t * DM);
  float4 xv[8];
#pragma unroll
  for (int j = 0; j < 8; ++j) xv[j] = xr[lane + 32 * j];

  // side output: bf16 copy of x for the v-projection GEMM
#pragma unroll
  for (int j = 0; j < 8; ++j) {
    union { unsigned short u[4]; uint2 q; } pk;
    pk.u[0] = f2bf(xv[j].x); pk.u[1] = f2bf(xv[j].y);
    pk.u[2] = f2bf(xv[j].z); pk.u[3] = f2bf(xv[j].w);
    *(uint2*)(xbf + t * DM + (size_t)(lane + 32 * j) * 4) = pk.q;
  }

  float th[12];
#pragma unroll
  for (int r = 0; r < 12; ++r) {
    float a = 0.0f;
#pragma unroll
    for (int j = 0; j < 8; ++j) {
      float4 wv = Wsh4[r * 256 + lane + 32 * j];
      a += xv[j].x * wv.x + xv[j].y * wv.y + xv[j].z * wv.z + xv[j].w * wv.w;
    }
#pragma unroll
    for (int off = 16; off; off >>= 1) a += __shfl_xor(a, off, 32);
    th[r] = a + (r < 6 ? bq[r] : bk[r - 6]);
  }
  run_circuit(th,     qs + t * DST, lane);
  run_circuit(th + 6, ks + t * DST, lane);
}

// ---------------------------------------------------------------------------
// K2: f32 -> bf16 (weights).
// ---------------------------------------------------------------------------
__global__ __launch_bounds__(256) void convert_bf16_kernel(
    const float* __restrict__ src, unsigned short* __restrict__ dst, int n4) {
  int idx = blockIdx.x * 256 + threadIdx.x;
  if (idx < n4) {
    float4 v = ((const float4*)src)[idx];
    union { unsigned short u[4]; uint2 q; } pk;
    pk.u[0] = f2bf(v.x); pk.u[1] = f2bf(v.y); pk.u[2] = f2bf(v.z); pk.u[3] = f2bf(v.w);
    *(uint2*)(dst + (size_t)idx * 4) = pk.q;
  }
}

// ---------------------------------------------------------------------------
// bf16 GEMM: C[m][n] = sum_k A[m][k]*Bt[n][k] (+bias). Macro tile 128x64,
// K-tile 32, double-buffered async LDS staging, 4 WMMAs per wave per K-step.
// ---------------------------------------------------------------------------
__global__ __launch_bounds__(256) void gemm_bf16_kernel(
    const unsigned short* __restrict__ A, const unsigned short* __restrict__ Bt,
    const float* __restrict__ bias, float* __restrict__ Cf,
    unsigned short* __restrict__ Cb,
    int M, int N, int K, long bStride, int rowsPerBatch, int transposedOut) {
  __shared__ __align__(16) unsigned short As[2][128][40];
  __shared__ __align__(16) unsigned short Bs[2][64][40];
  const int tid  = threadIdx.x;
  const int lane = tid & 31;
  const int w    = tid >> 5;
  const int m0   = blockIdx.y * 128;
  const int n0   = blockIdx.x * 64;
  const unsigned short* Bp = Bt + (bStride ? (long)(m0 / rowsPerBatch) * bStride : 0L);

  const int mt0 = (w & 3) * 2;
  const int nt0 = (w >> 2) * 2;
  v8f acc00 = {}, acc01 = {}, acc10 = {}, acc11 = {};

  const int amA = mt0 * 16 + (lane & 15);
  const int akb = (lane >> 4) * 8;
  const int bnA = nt0 * 16 + (lane & 15);
  const int bkb = (lane >> 4) * 8;

  auto stage = [&](int buf, int k0) {
#pragma unroll
    for (int it = 0; it < 2; ++it) {        // A: 128 rows x 4 x 16B
      int l = tid + it * 256, r = l >> 2, c = l & 3;
      g2l_16B(&As[buf][r][c * 8], A + (size_t)(m0 + r) * K + k0 + c * 8);
    }
    {                                       // B: 64 rows x 4 x 16B
      int r = tid >> 2, c = tid & 3;
      g2l_16B(&Bs[buf][r][c * 8], Bp + (size_t)(n0 + r) * K + k0 + c * 8);
    }
  };

  stage(0, 0);
  for (int k0 = 0; k0 < K; k0 += 32) {
    async_fence();
    __syncthreads();
    const int cur = (k0 >> 5) & 1;
    if (k0 + 32 < K) stage(cur ^ 1, k0 + 32);

    FragBF a0, a1, b0, b1;
    a0.h[0] = *(const v8bf*)&As[cur][amA][akb];
    a0.h[1] = *(const v8bf*)&As[cur][amA][16 + akb];
    a1.h[0] = *(const v8bf*)&As[cur][amA + 16][akb];
    a1.h[1] = *(const v8bf*)&As[cur][amA + 16][16 + akb];
    b0.h[0] = *(const v8bf*)&Bs[cur][bnA][bkb];
    b0.h[1] = *(const v8bf*)&Bs[cur][bnA][16 + bkb];
    b1.h[0] = *(const v8bf*)&Bs[cur][bnA + 16][bkb];
    b1.h[1] = *(const v8bf*)&Bs[cur][bnA + 16][16 + bkb];
    acc00 = __builtin_amdgcn_wmma_f32_16x16x32_bf16(false, a0.v, false, b0.v, (short)0, acc00, false, false);
    acc01 = __builtin_amdgcn_wmma_f32_16x16x32_bf16(false, a0.v, false, b1.v, (short)0, acc01, false, false);
    acc10 = __builtin_amdgcn_wmma_f32_16x16x32_bf16(false, a1.v, false, b0.v, (short)0, acc10, false, false);
    acc11 = __builtin_amdgcn_wmma_f32_16x16x32_bf16(false, a1.v, false, b1.v, (short)0, acc11, false, false);
  }

  const int cn    = lane & 15;
  const int rbase = (lane >> 4) * 8;
#pragma unroll
  for (int tm = 0; tm < 2; ++tm) {
#pragma unroll
    for (int tn = 0; tn < 2; ++tn) {
      const v8f& acc = tm ? (tn ? acc11 : acc10) : (tn ? acc01 : acc00);
      int gn = n0 + (nt0 + tn) * 16 + cn;
      float bb = bias ? bias[gn] : 0.0f;
#pragma unroll
      for (int i = 0; i < 8; ++i) {
        int gm = m0 + (mt0 + tm) * 16 + rbase + i;
        float val = acc[i] + bb;
        if (Cf) Cf[(size_t)gm * N + gn] = val;
        if (Cb) {
          unsigned short h = f2bf(val);
          if (transposedOut) {
            int b = gm / rowsPerBatch, s = gm % rowsPerBatch;
            Cb[((size_t)b * N + gn) * rowsPerBatch + s] = h;
          } else {
            Cb[(size_t)gm * N + gn] = h;
          }
        }
      }
    }
  }
}

// ---------------------------------------------------------------------------
// K4: scores = (q.k)^2 centered/scaled/masked. f32 operands, K=64,
// exact-f32 via V_WMMA_F32_16X16X4_F32 (16 K-steps, 2 tiles per wave).
// ---------------------------------------------------------------------------
__global__ __launch_bounds__(256) void scores_kernel(
    const float* __restrict__ qs, const float* __restrict__ ks,
    const unsigned char* __restrict__ mask, const float* __restrict__ lsc,
    float* __restrict__ attn) {
  __shared__ __align__(16) float As[64][68];
  __shared__ __align__(16) float Bs[64][68];
  const int tid  = threadIdx.x;
  const int lane = tid & 31;
  const int w    = tid >> 5;
  const int b    = blockIdx.z;
  const int i0   = blockIdx.y * 64;
  const int j0   = blockIdx.x * 64;
  const float* Aq = qs + ((size_t)b * SEQ + i0) * DST;
  const float* Bk = ks + ((size_t)b * SEQ + j0) * DST;

#pragma unroll
  for (int it = 0; it < 4; ++it) {          // 64 rows x 16 x 16B each
    int l = tid + it * 256, r = l >> 4, c = l & 15;
    g2l_16B(&As[r][c * 4], Aq + (size_t)r * DST + c * 4);
    g2l_16B(&Bs[r][c * 4], Bk + (size_t)r * DST + c * 4);
  }
  async_fence();
  __syncthreads();

  const int mt  = w & 3;
  const int nt0 = (w >> 2) * 2;
  const int am  = mt * 16 + (lane & 15);
  const int bn  = nt0 * 16 + (lane & 15);
  const int kh  = (lane >> 4) * 2;          // f32 A/B frag: cols 4k + kh + {0,1}
  v8f acc0 = {}, acc1 = {};
#pragma unroll
  for (int kk = 0; kk < 16; ++kk) {
    v2f a  = *(const v2f*)&As[am][kk * 4 + kh];
    v2f b0 = *(const v2f*)&Bs[bn][kk * 4 + kh];
    v2f b1 = *(const v2f*)&Bs[bn + 16][kk * 4 + kh];
    acc0 = __builtin_amdgcn_wmma_f32_16x16x4_f32(false, a, false, b0, (short)0, acc0, false, false);
    acc1 = __builtin_amdgcn_wmma_f32_16x16x4_f32(false, a, false, b1, (short)0, acc1, false, false);
  }

  const float esc = __expf(lsc[0]);
  const int cn    = lane & 15;
  const int rbase = (lane >> 4) * 8;
#pragma unroll
  for (int i = 0; i < 8; ++i) {
    int gi = i0 + mt * 16 + rbase + i;
#pragma unroll
    for (int tnt = 0; tnt < 2; ++tnt) {
      int gj = j0 + (tnt + nt0) * 16 + cn;
      float ov = tnt ? acc1[i] : acc0[i];
      float val = (ov * ov - 1.0f / DST) * esc;
      if (!mask[(size_t)b * SEQ + gj]) val = -1000000000.0f;
      attn[((size_t)b * SEQ + gi) * SEQ + gj] = val;
    }
  }
}

// ---------------------------------------------------------------------------
// K5: row softmax in place (f32) + bf16 copy for the attn@v GEMM.
// ---------------------------------------------------------------------------
__global__ __launch_bounds__(256) void softmax_kernel(
    float* __restrict__ attn, unsigned short* __restrict__ attnbf,
    const unsigned char* __restrict__ mask) {
  const int row = blockIdx.x;
  float* p = attn + (size_t)row * SEQ;
  unsigned short* pb = attnbf + (size_t)row * SEQ;
  const int tid = threadIdx.x;
  __shared__ float red[256];

  float v[8];
  float mx = -3.4e38f;
#pragma unroll
  for (int j = 0; j < 8; ++j) { v[j] = p[tid + 256 * j]; mx = fmaxf(mx, v[j]); }
  red[tid] = mx; __syncthreads();
  for (int s = 128; s; s >>= 1) { if (tid < s) red[tid] = fmaxf(red[tid], red[tid + s]); __syncthreads(); }
  mx = red[0]; __syncthreads();

  float sum = 0.0f;
#pragma unroll
  for (int j = 0; j < 8; ++j) { v[j] = __expf(v[j] - mx); sum += v[j]; }
  red[tid] = sum; __syncthreads();
  for (int s = 128; s; s >>= 1) { if (tid < s) red[tid] += red[tid + s]; __syncthreads(); }
  sum = red[0];

  const float scale = (mask[row] ? 1.0f : 0.0f) / sum;
#pragma unroll
  for (int j = 0; j < 8; ++j) {
    float o = v[j] * scale;
    p[tid + 256 * j]  = o;
    pb[tid + 256 * j] = f2bf(o);
  }
}

// ---------------------------------------------------------------------------
extern "C" void kernel_launch(void* const* d_in, const int* in_sizes, int n_in,
                              void* d_out, int out_size, void* d_ws, size_t ws_size,
                              hipStream_t stream) {
  (void)in_sizes; (void)n_in; (void)out_size; (void)ws_size;
  const float* x   = (const float*)d_in[0];
  const unsigned char* mask = (const unsigned char*)d_in[1];
  const float* Wq  = (const float*)d_in[2];
  const float* bq  = (const float*)d_in[3];
  const float* Wk  = (const float*)d_in[4];
  const float* bk  = (const float*)d_in[5];
  const float* Wv  = (const float*)d_in[6];
  const float* bv  = (const float*)d_in[7];
  const float* Wo  = (const float*)d_in[8];
  const float* bo  = (const float*)d_in[9];
  const float* lsc = (const float*)d_in[10];

  const int M = BATCH * SEQ;
  float* out  = (float*)d_out;
  float* attn = out + (size_t)M * DM;                 // attention output region

  float* qs_f = (float*)d_ws;                                      //  4 MB
  float* ks_f = qs_f + (size_t)M * DST;                            //  4 MB
  unsigned short* x_bf     = (unsigned short*)(ks_f + (size_t)M * DST); // 32 MB
  unsigned short* Wv_bf    = x_bf     + (size_t)M * DM;            //  2 MB
  unsigned short* Wo_bf    = Wv_bf    + (size_t)DM * DM;           //  2 MB
  unsigned short* vT_bf    = Wo_bf    + (size_t)DM * DM;           // 32 MB [b][n][s]
  unsigned short* attnV_bf = vT_bf    + (size_t)BATCH * DM * SEQ;  // 32 MB
  unsigned short* attn_bf  = attnV_bf + (size_t)M * DM;            // 64 MB

  encode_kernel<<<dim3(M / 8), dim3(256), 0, stream>>>(x, Wq, bq, Wk, bk,
                                                       qs_f, ks_f, x_bf);
  convert_bf16_kernel<<<dim3(DM * DM / 1024), dim3(256), 0, stream>>>(Wv, Wv_bf, DM * DM / 4);
  convert_bf16_kernel<<<dim3(DM * DM / 1024), dim3(256), 0, stream>>>(Wo, Wo_bf, DM * DM / 4);

  // vT_bf = bf16((x @ Wv.T + bv)) stored transposed per batch
  gemm_bf16_kernel<<<dim3(DM / 64, M / 128), dim3(256), 0, stream>>>(
      x_bf, Wv_bf, bv, nullptr, vT_bf, M, DM, DM, 0L, SEQ, 1);

  scores_kernel<<<dim3(SEQ / 64, SEQ / 64, BATCH), dim3(256), 0, stream>>>(
      qs_f, ks_f, mask, lsc, attn);

  softmax_kernel<<<dim3(M), dim3(256), 0, stream>>>(attn, attn_bf, mask);

  // attnV_bf = bf16(attention @ v)   (Bt = vT_bf per batch)
  gemm_bf16_kernel<<<dim3(DM / 64, M / 128), dim3(256), 0, stream>>>(
      attn_bf, vT_bf, nullptr, nullptr, attnV_bf, M, DM, SEQ, (long)DM * SEQ, SEQ, 0);

  // out = attnV @ Wo.T + bo  (f32)
  gemm_bf16_kernel<<<dim3(DM / 64, M / 128), dim3(256), 0, stream>>>(
      attnV_bf, Wo_bf, bo, out, nullptr, M, DM, DM, 0L, SEQ, 0);
}
